// AECombinedLoss_23613730193971
// MI455X (gfx1250) — matrix operand order repaired and checked by reading, not compiled, not run on hardware
//
#include <hip/hip_runtime.h>

typedef __attribute__((ext_vector_type(16))) _Float16     v16h;
typedef __attribute__((ext_vector_type(8)))  _Float16     v8h;
typedef __attribute__((ext_vector_type(8)))  float        v8f;
typedef __attribute__((ext_vector_type(4)))  float        f32x4;
typedef __attribute__((ext_vector_type(4)))  unsigned int u32x4;
typedef __attribute__((ext_vector_type(8)))  int          i32x8;
typedef __attribute__((ext_vector_type(4)))  int          i32x4;

#define IMG     512
#define PLANES  (64*3)
#define RROWS   48
#define RSTRIDE 68          // f32 LDS row stride (272B): TDM writes 64 dwords + 4 pad dwords
#define HT_STR  40          // Ht row stride in halfs (80B)
#define NTOT    50331648.0f // 64*3*512*512 == 3*2^24, exact in f32
#define C1F 4.0e-4f         // (0.01*2)^2
#define C2F 3.6e-3f         // (0.03*2)^2

__device__ __forceinline__ float waveSum32(float v) {
    #pragma unroll
    for (int m = 16; m >= 1; m >>= 1) v += __shfl_xor(v, m, 32);
    return v;
}

// Border-block (or no-TDM) staging: predicated loads, zero outside image.
// Writes rows 0..47, cols 0..47 of the strided LDS tile (cols>=48 never read).
__device__ __forceinline__ void stage_valu(float* Xf, float* Yf,
                                           const float* __restrict__ xg,
                                           const float* __restrict__ yg,
                                           int r0, int c0, int tid) {
    for (int it = 0; it < 18; ++it) {           // 48*48 / 128 == 18 exactly
        int idx = it * 128 + tid;
        int r = idx / 48, c = idx % 48;
        int gr = r0 + r, gc = c0 + c;
        bool v = (gr >= 0) && (gr < IMG) && (gc >= 0) && (gc < IMG);
        float xv = 0.f, yv = 0.f;
        if (v) { int off = gr * IMG + gc; xv = xg[off]; yv = yg[off]; }
        Xf[r * RSTRIDE + c] = xv;
        Yf[r * RSTRIDE + c] = yv;
    }
}

__global__ void __launch_bounds__(128)
ssim_mse_wmma_kernel(const float* __restrict__ recon,
                     const float* __restrict__ orig,
                     float* __restrict__ accum)
{
    __shared__ __align__(16) float    Xf[RROWS * RSTRIDE];
    __shared__ __align__(16) float    Yf[RROWS * RSTRIDE];
    __shared__ __align__(16) _Float16 Hts[4][5][16][HT_STR];
    __shared__ float red[2][4];

    const int tid  = threadIdx.x;
    const int w    = tid >> 5;
    const int lane = tid & 31;
    const int p  = blockIdx.x >> 8;       // plane (256 32x32 groups per plane)
    const int g  = blockIdx.x & 255;
    const int by = g >> 4, bx = g & 15;
    const int r0 = by * 32 - 5, c0 = bx * 32 - 5;

    const float* xg = recon + (size_t)p * (IMG * IMG);
    const float* yg = orig  + (size_t)p * (IMG * IMG);

    const bool interior = (by >= 1) && (by <= 14) && (bx >= 1) && (bx <= 14);

    // ---- stage 48-row f32 halo tile into LDS ----
#if __has_builtin(__builtin_amdgcn_tensor_load_to_lds) && __has_builtin(__builtin_amdgcn_s_wait_tensorcnt)
    if (interior) {
        // TDM DMA: 48x64 f32 tile, LDS row stride padded 64->68 dwords.
        if (w == 0) {
            unsigned long long gax = (unsigned long long)(const void*)(xg + (r0 * IMG + c0));
            unsigned long long gay = (unsigned long long)(const void*)(yg + (r0 * IMG + c0));
            unsigned ldsX = (unsigned)(unsigned long long)(void*)Xf;
            unsigned ldsY = (unsigned)(unsigned long long)(void*)Yf;
            // D# group1 (ISA §8.4): data_size=2(4B), pad_enable=1, pad_interval=5(64dw),
            // pad_amount=3(4dw); tensor_dim0=512, tensor_dim1=512; tile 64x48;
            // dim0_stride=512, dim1_stride=512*512.
            i32x8 g1 = { 0x07520000, 0x02000000, 0x02000000, 0x00400000,
                         48, 512, 0, 4 };
            i32x4 z4 = { 0, 0, 0, 0 };
            i32x8 z8 = { 0, 0, 0, 0, 0, 0, 0, 0 };
            // D# group0 (ISA §8.3): count=1 | lds_addr | global_addr | type=2.
            u32x4 g0x = { 1u, ldsX, (unsigned)gax,
                          (unsigned)(gax >> 32) | 0x80000000u };
            u32x4 g0y = { 1u, ldsY, (unsigned)gay,
                          (unsigned)(gay >> 32) | 0x80000000u };
            // amdgpu-toolchain (clang-23) 6-arg form.
            __builtin_amdgcn_tensor_load_to_lds(g0x, g1, z4, z4, z8, 0);
            __builtin_amdgcn_tensor_load_to_lds(g0y, g1, z4, z4, z8, 0);
            __builtin_amdgcn_s_wait_tensorcnt(0);
        }
    } else {
        stage_valu(Xf, Yf, xg, yg, r0, c0, tid);
    }
#else
    stage_valu(Xf, Yf, xg, yg, r0, c0, tid);
#endif
    __syncthreads();

    // ---- per-wave quadrant of the 32x32 output region ----
    const int qy = (w >> 1) & 1, qx = w & 1;
    const int nlo = lane & 15;
    const int hi  = lane >> 4;

    // normalized 1D Gaussian (sigma=1.5, 11 taps)
    float gw[11];
    {
        float s = 0.f;
        #pragma unroll
        for (int u = 0; u < 11; ++u) {
            float d = (float)(u - 5);
            gw[u] = __expf(-(d * d) / 4.5f);
            s += gw[u];
        }
        float inv = 1.f / s;
        #pragma unroll
        for (int u = 0; u < 11; ++u) gw[u] *= inv;
    }

    // horizontal B constant: Wh[k][j] = g[k-j]   (B 32x16 f16 layout)
    v16h whb;
    #pragma unroll
    for (int e = 0; e < 16; ++e) {
        int k = e + hi * 8 + ((e >= 8) ? 8 : 0);
        int d = k - nlo;
        whb[e] = (_Float16)((d >= 0 && d < 11) ? gw[d] : 0.f);
    }
    // vertical A constant: Wv[i][k] = g[k-i]     (A 16x32 f16 layout)
    v16h wva;
    #pragma unroll
    for (int e = 0; e < 16; ++e) {
        int k = hi * 16 + e;
        int d = k - nlo;
        wva[e] = (_Float16)((d >= 0 && d < 11) ? gw[d] : 0.f);
    }

    // ---- horizontal pass: 2 row-blocks x 5 maps, one WMMA each ----
    #pragma unroll
    for (int bb = 0; bb < 2; ++bb) {
        const float* xrow = &Xf[(qy * 16 + bb * 16 + nlo) * RSTRIDE + qx * 16];
        const float* yrow = &Yf[(qy * 16 + bb * 16 + nlo) * RSTRIDE + qx * 16];
        f32x4 x0 = *(const f32x4*)(xrow + 8 * hi);
        f32x4 x1 = *(const f32x4*)(xrow + 8 * hi + 4);
        f32x4 x2 = *(const f32x4*)(xrow + 16 + 8 * hi);
        f32x4 x3 = *(const f32x4*)(xrow + 16 + 8 * hi + 4);
        f32x4 y0 = *(const f32x4*)(yrow + 8 * hi);
        f32x4 y1 = *(const f32x4*)(yrow + 8 * hi + 4);
        f32x4 y2 = *(const f32x4*)(yrow + 16 + 8 * hi);
        f32x4 y3 = *(const f32x4*)(yrow + 16 + 8 * hi + 4);
        v16h xa, ya;
        #pragma unroll
        for (int e = 0; e < 4; ++e) {
            xa[e]      = (_Float16)x0[e];
            xa[4 + e]  = (_Float16)x1[e];
            xa[8 + e]  = (_Float16)x2[e];
            xa[12 + e] = (_Float16)x3[e];
            ya[e]      = (_Float16)y0[e];
            ya[4 + e]  = (_Float16)y1[e];
            ya[8 + e]  = (_Float16)y2[e];
            ya[12 + e] = (_Float16)y3[e];
        }

        v16h amap[5];
        amap[0] = xa;
        amap[1] = ya;
        amap[2] = xa * xa;    // packed f16 muls
        amap[3] = ya * ya;
        amap[4] = xa * ya;

        #pragma unroll
        for (int m5 = 0; m5 < 5; ++m5) {
            v8f cz = {};
            v8f hcol = __builtin_amdgcn_wmma_f32_16x16x32_f16(
                false, amap[m5], false, whb, (short)0, cz, false, false);
            v8h hh;
            #pragma unroll
            for (int e = 0; e < 8; ++e) hh[e] = (_Float16)hcol[e];
            // lane owns H[bb*16 + 8*hi + e][nlo]; store transposed -> one ds_store_b128
            *(v8h*)(&Hts[w][m5][nlo][bb * 16 + 8 * hi]) = hh;
        }
    }

    // ---- vertical pass: one WMMA per map ----
    v8f dmap[5];
    #pragma unroll
    for (int m5 = 0; m5 < 5; ++m5) {
        const _Float16* hrow = &Hts[w][m5][nlo][16 * hi];
        v8h bl = *(const v8h*)(hrow);
        v8h bh = *(const v8h*)(hrow + 8);
        v16h b = __builtin_shufflevector(bl, bh, 0,1,2,3,4,5,6,7,8,9,10,11,12,13,14,15);
        v8f cz = {};
        dmap[m5] = __builtin_amdgcn_wmma_f32_16x16x32_f16(
            false, wva, false, b, (short)0, cz, false, false);
    }

    // ---- SSIM map + exact-f32 MSE over this lane's 8 pixels ----
    float ssimAcc = 0.f, mseAcc = 0.f;
    #pragma unroll
    for (int e = 0; e < 8; ++e) {
        float mu1 = dmap[0][e], mu2 = dmap[1][e];
        float ex2 = dmap[2][e], ey2 = dmap[3][e], exy = dmap[4][e];
        float mu1s = mu1 * mu1, mu2s = mu2 * mu2, mu12 = mu1 * mu2;
        float s1  = ex2 - mu1s, s2 = ey2 - mu2s, s12 = exy - mu12;
        float num = (2.f * mu12 + C1F) * (2.f * s12 + C2F);
        float den = (mu1s + mu2s + C1F) * (s1 + s2 + C2F);
        ssimAcc += num / den;

        int rr = qy * 16 + (e + 8 * hi) + 5;   // core pixel, counted exactly once
        int cc = qx * 16 + nlo + 5;
        float xv = Xf[rr * RSTRIDE + cc];
        float yv = Yf[rr * RSTRIDE + cc];
        float d = xv - yv;
        mseAcc += d * d;
    }

    // ---- reduce: wave -> block -> global atomics ----
    mseAcc  = waveSum32(mseAcc);
    ssimAcc = waveSum32(ssimAcc);
    if (lane == 0) { red[0][w] = mseAcc; red[1][w] = ssimAcc; }
    __syncthreads();
    if (tid == 0) {
        float m = 0.f, s = 0.f;
        #pragma unroll
        for (int i = 0; i < 4; ++i) { m += red[0][i]; s += red[1][i]; }
        atomicAdd(&accum[0], m);
        atomicAdd(&accum[1], s);
    }
}

__global__ void loss_init_kernel(float* accum) {
    accum[0] = 0.f;
    accum[1] = 0.f;
}

__global__ void loss_final_kernel(const float* __restrict__ accum,
                                  float* __restrict__ out) {
    float mse  = accum[0] / NTOT;
    float ssim = accum[1] / NTOT;
    out[0] = 0.7f * mse + 0.3f * (1.f - ssim);
}

extern "C" void kernel_launch(void* const* d_in, const int* in_sizes, int n_in,
                              void* d_out, int out_size, void* d_ws, size_t ws_size,
                              hipStream_t stream) {
    const float* recon = (const float*)d_in[0];
    const float* orig  = (const float*)d_in[1];
    float* out   = (float*)d_out;
    float* accum = (float*)d_ws;

    loss_init_kernel<<<1, 1, 0, stream>>>(accum);

    const int nBlocks = PLANES * 256;   // 49,152 blocks of 128 threads (4 tiles each)
    ssim_mse_wmma_kernel<<<nBlocks, 128, 0, stream>>>(recon, orig, accum);

    loss_final_kernel<<<1, 1, 0, stream>>>(accum, out);
}
